// GraphTransformerLayer_20993800143386
// MI455X (gfx1250) — matrix operand order repaired and checked by reading, not compiled, not run on hardware
//
#include <hip/hip_runtime.h>
#include <hip/hip_bf16.h>
#include <math.h>

#define N_NODES 50000
#define N_EDGES 800000
#define HID 128
#define HEADS 8
#define DKH 16
#define FF 512

typedef __attribute__((ext_vector_type(16))) _Float16 v16h;
typedef __attribute__((ext_vector_type(8)))  _Float16 v8h;
typedef __attribute__((ext_vector_type(8)))  float    v8f;

static __device__ __forceinline__ v16h frag_cat(v8h lo, v8h hi) {
  return __builtin_shufflevector(lo, hi, 0,1,2,3,4,5,6,7,8,9,10,11,12,13,14,15);
}

// A fragment (16x32 f16, M x K) from an LDS row-major tile with leading dim ldk.
// Per ISA 7.12.2: lanes 0-15 hold row M=lane with K {0..7, 16..23}; lanes 16-31 same row, K {8..15, 24..31}.
static __device__ __forceinline__ v16h ldsA_frag(const _Float16* As, int lane, int kc, int ldk) {
  int row = lane & 15;
  int k0  = kc * 32 + ((lane & 16) ? 8 : 0);
  const _Float16* p = As + row * ldk + k0;
  v8h lo = *(const v8h*)(p);
  v8h hi = *(const v8h*)(p + 16);
  return frag_cat(lo, hi);
}

// B fragment (32x16 f16, K x N) from a column-major (transposed) f16 weight WT[n][k], leading dim ldk.
// Lanes 0-15: column N=lane, K {0..7, 16..23} ; lanes 16-31: same column, K {8..15, 24..31} (matches A).
static __device__ __forceinline__ v16h glbB_frag(const _Float16* WT, int colg, int lane, int kc, int ldk) {
  int k0 = kc * 32 + ((lane & 16) ? 8 : 0);
  const _Float16* p = WT + (size_t)colg * ldk + k0;
  v8h lo = *(const v8h*)(p);
  v8h hi = *(const v8h*)(p + 16);
  return frag_cat(lo, hi);
}

#define WMMA_F16(a, b, c) \
  __builtin_amdgcn_wmma_f32_16x16x32_f16(false, (a), false, (b), (short)0, (c), false, false)

// Async global->LDS 16B copy (ASYNCcnt-tracked, LDS written without touching VGPRs).
// LDS byte offset = low 32 bits of the generic pointer (ISA 10.2).
static __device__ __forceinline__ void async_lds_b128(void* lds_dst, const void* gsrc) {
  unsigned lds_off = (unsigned)(size_t)lds_dst;
  asm volatile("global_load_async_to_lds_b128 %0, %1, off"
               :: "v"(lds_off), "v"(gsrc)
               : "memory");
}
static __device__ __forceinline__ void wait_async0() {
  asm volatile("s_wait_asynccnt 0x0" ::: "memory");
}

// ---------------------------------------------------------------------------
// Weight transpose + f32 -> f16 convert:  dst[n*K + k] = src[k*Ncol + n]
__global__ void transpose_f16_kernel(_Float16* __restrict__ dst, const float* __restrict__ src,
                                     int K, int Ncol) {
  int total = K * Ncol;
  int stride = gridDim.x * blockDim.x;
  for (int i = blockIdx.x * blockDim.x + threadIdx.x; i < total; i += stride) {
    int n = i / K;
    int k = i - n * K;
    dst[i] = (_Float16)src[(size_t)k * Ncol + n];
  }
}

// Elementwise f32 -> f16 mirror (x16 feed for async A staging)
__global__ void convert_f16_kernel(_Float16* __restrict__ dst, const float* __restrict__ src,
                                   int total) {
  int stride = gridDim.x * blockDim.x;
  for (int i = blockIdx.x * blockDim.x + threadIdx.x; i < total; i += stride)
    dst[i] = (_Float16)src[i];
}

// Zero agg, init per-(node,head) max to -inf-ish and sum to 0.
__global__ void init_ws_kernel(float* __restrict__ agg, float* __restrict__ nmax,
                               float* __restrict__ nsum) {
  int stride = gridDim.x * blockDim.x;
  int gid = blockIdx.x * blockDim.x + threadIdx.x;
  for (int i = gid; i < N_NODES * HID; i += stride) agg[i] = 0.0f;
  for (int i = gid; i < N_NODES * HEADS; i += stride) { nmax[i] = -3.0e38f; nsum[i] = 0.0f; }
}

// ---------------------------------------------------------------------------
// Fused QKV GEMM: [N,128] x [128, 384] (+bias) -> Q,K,V  (f32 outs, f16 WMMA math)
__global__ __launch_bounds__(256) void qkv_gemm_kernel(
    const _Float16* __restrict__ x16, const _Float16* __restrict__ WqkvT,
    const float* __restrict__ bq, const float* __restrict__ bk, const float* __restrict__ bv,
    float* __restrict__ Q, float* __restrict__ K, float* __restrict__ V) {
  __shared__ _Float16 As[16 * HID];
  int tid = threadIdx.x, lane = tid & 31, wave = tid >> 5;
  int rowbase = blockIdx.x * 16;
  // Stage the 16x128 f16 A tile via async global->LDS (one b128 per thread).
  async_lds_b128(&As[tid * 8], x16 + (size_t)rowbase * HID + tid * 8);
  wait_async0();
  __syncthreads();

  // A fragments shared by every column tile this wave touches.
  v16h a0 = ldsA_frag(As, lane, 0, HID);
  v16h a1 = ldsA_frag(As, lane, 1, HID);
  v16h a2 = ldsA_frag(As, lane, 2, HID);
  v16h a3 = ldsA_frag(As, lane, 3, HID);

  int col = lane & 15;
  int rb  = (lane & 16) ? 8 : 0;
#pragma unroll
  for (int g = 0; g < 3; g++) {
    int ct = wave + 8 * g;  // 0..23 across [Wq|Wk|Wv]
    float* outp; const float* bias;
    if (g == 0)      { outp = Q; bias = bq; }
    else if (g == 1) { outp = K; bias = bk; }
    else             { outp = V; bias = bv; }
    int cgB = ct * 16 + col;         // column in 384-wide WqkvT
    int cgO = (ct & 7) * 16 + col;   // column in 128-wide output
    float bb = bias[cgO];
    v8f acc;
#pragma unroll
    for (int r = 0; r < 8; r++) acc[r] = bb;
    acc = WMMA_F16(a0, glbB_frag(WqkvT, cgB, lane, 0, HID), acc);
    acc = WMMA_F16(a1, glbB_frag(WqkvT, cgB, lane, 1, HID), acc);
    acc = WMMA_F16(a2, glbB_frag(WqkvT, cgB, lane, 2, HID), acc);
    acc = WMMA_F16(a3, glbB_frag(WqkvT, cgB, lane, 3, HID), acc);
#pragma unroll
    for (int r = 0; r < 8; r++)
      outp[(size_t)(rowbase + rb + r) * HID + cgO] = acc[r];
  }
}

// ---------------------------------------------------------------------------
// Edge attention, pass 1: per-(edge,head) score + float atomic max per target.
static __device__ __forceinline__ void atomicMaxF(float* addr, float v) {
  if (v >= 0.0f) atomicMax((int*)addr, __float_as_int(v));
  else           atomicMin((unsigned int*)addr, __float_as_uint(v));
}

__global__ void attn_score_kernel(const float* __restrict__ Q, const float* __restrict__ K,
                                  const int* __restrict__ ei, float* __restrict__ esc,
                                  float* __restrict__ nmax) {
  int gid = blockIdx.x * blockDim.x + threadIdx.x;
  if (gid >= N_EDGES * HEADS) return;
  int e = gid >> 3, h = gid & 7;
  int src = ei[e], tgt = ei[N_EDGES + e];
  const float* q = Q + (size_t)tgt * HID + h * DKH;
  const float* k = K + (size_t)src * HID + h * DKH;
  float s = 0.0f;
#pragma unroll
  for (int i = 0; i < DKH; i++) s += q[i] * k[i];
  s *= 0.25f;  // 1/sqrt(16)
  esc[gid] = s;
  atomicMaxF(&nmax[tgt * HEADS + h], s);
}

// pass 2: exp(score - max) + atomic sum per target
__global__ void attn_exp_kernel(const int* __restrict__ ei, float* __restrict__ esc,
                                const float* __restrict__ nmax, float* __restrict__ nsum) {
  int gid = blockIdx.x * blockDim.x + threadIdx.x;
  if (gid >= N_EDGES * HEADS) return;
  int e = gid >> 3, h = gid & 7;
  int tgt = ei[N_EDGES + e];
  float ex = __expf(esc[gid] - nmax[tgt * HEADS + h]);
  esc[gid] = ex;
  atomicAdd(&nsum[tgt * HEADS + h], ex);
}

// pass 3: alpha * V[src] scattered into agg[tgt] with fp32 atomics (resolved in L2)
__global__ void attn_agg_kernel(const int* __restrict__ ei, const float* __restrict__ esc,
                                const float* __restrict__ nsum, const float* __restrict__ V,
                                float* __restrict__ agg) {
  int gid = blockIdx.x * blockDim.x + threadIdx.x;
  if (gid >= N_EDGES * HEADS) return;
  int e = gid >> 3, h = gid & 7;
  int src = ei[e], tgt = ei[N_EDGES + e];
  float alpha = esc[gid] / (nsum[tgt * HEADS + h] + 1e-16f);
  const float* v = V + (size_t)src * HID + h * DKH;
  float* o = agg + (size_t)tgt * HID + h * DKH;
#pragma unroll
  for (int i = 0; i < DKH; i++) atomicAdd(&o[i], alpha * v[i]);
}

// ---------------------------------------------------------------------------
// O-projection + residual + LayerNorm1:  h = LN(x + agg @ Wo + bo); also emits h16.
__global__ __launch_bounds__(256) void o_ln_kernel(
    const float* __restrict__ agg, const _Float16* __restrict__ WoT,
    const float* __restrict__ bo, const float* __restrict__ x,
    const float* __restrict__ g, const float* __restrict__ b,
    float* __restrict__ hout, _Float16* __restrict__ h16out) {
  __shared__ _Float16 As[16 * HID];
  __shared__ float hbuf[16 * HID];
  int tid = threadIdx.x, lane = tid & 31, wave = tid >> 5;
  int rowbase = blockIdx.x * 16;
  {
    int idx = tid * 8;  // agg is f32 (atomic-produced): convert on stage-in
    const float* s = agg + (size_t)rowbase * HID + idx;
#pragma unroll
    for (int i = 0; i < 8; i++) As[idx + i] = (_Float16)s[i];
  }
  __syncthreads();
  int col = lane & 15, rb = (lane & 16) ? 8 : 0;
  int cg = wave * 16 + col;  // one col tile per wave, 8 waves cover 128 cols
  float bb = bo[cg];
  v8f acc;
#pragma unroll
  for (int r = 0; r < 8; r++) acc[r] = bb;
#pragma unroll
  for (int kc = 0; kc < 4; kc++)
    acc = WMMA_F16(ldsA_frag(As, lane, kc, HID), glbB_frag(WoT, cg, lane, kc, HID), acc);
#pragma unroll
  for (int r = 0; r < 8; r++) {
    int row = rb + r;
    hbuf[row * HID + cg] = acc[r] + x[(size_t)(rowbase + row) * HID + cg];
  }
  __syncthreads();
  if (tid < 16) {
    const float* hr = &hbuf[tid * HID];
    float mu = 0.0f;
    for (int i = 0; i < HID; i++) mu += hr[i];
    mu *= (1.0f / HID);
    float var = 0.0f;
    for (int i = 0; i < HID; i++) { float d = hr[i] - mu; var += d * d; }
    var *= (1.0f / HID);
    float rs = rsqrtf(var + 1e-5f);
    float* o = hout + (size_t)(rowbase + tid) * HID;
    _Float16* o16 = h16out + (size_t)(rowbase + tid) * HID;
    for (int i = 0; i < HID; i++) {
      float v = (hr[i] - mu) * rs * g[i] + b[i];
      o[i] = v;
      o16[i] = (_Float16)v;
    }
  }
}

// ---------------------------------------------------------------------------
// FFN1: ff1 = relu(h @ W1 + b1), f16 output (feeds WMMA A of FFN2 directly)
__global__ __launch_bounds__(256) void ffn1_kernel(
    const _Float16* __restrict__ h16, const _Float16* __restrict__ W1T,
    const float* __restrict__ b1, _Float16* __restrict__ ff1) {
  __shared__ _Float16 As[16 * HID];
  int tid = threadIdx.x, lane = tid & 31, wave = tid >> 5;
  int rowbase = blockIdx.x * 16;
  async_lds_b128(&As[tid * 8], h16 + (size_t)rowbase * HID + tid * 8);
  wait_async0();
  __syncthreads();
  v16h a0 = ldsA_frag(As, lane, 0, HID);
  v16h a1 = ldsA_frag(As, lane, 1, HID);
  v16h a2 = ldsA_frag(As, lane, 2, HID);
  v16h a3 = ldsA_frag(As, lane, 3, HID);
  int col = lane & 15, rb = (lane & 16) ? 8 : 0;
#pragma unroll
  for (int g = 0; g < 4; g++) {
    int ct = wave + 8 * g;  // 0..31 covers 512 cols
    int cg = ct * 16 + col;
    float bb = b1[cg];
    v8f acc;
#pragma unroll
    for (int r = 0; r < 8; r++) acc[r] = bb;
    acc = WMMA_F16(a0, glbB_frag(W1T, cg, lane, 0, HID), acc);
    acc = WMMA_F16(a1, glbB_frag(W1T, cg, lane, 1, HID), acc);
    acc = WMMA_F16(a2, glbB_frag(W1T, cg, lane, 2, HID), acc);
    acc = WMMA_F16(a3, glbB_frag(W1T, cg, lane, 3, HID), acc);
#pragma unroll
    for (int r = 0; r < 8; r++)
      ff1[(size_t)(rowbase + rb + r) * FF + cg] = (_Float16)fmaxf(acc[r], 0.0f);
  }
}

// FFN2 + residual + LayerNorm2: out = LN(h + ff1 @ W2 + b2)
__global__ __launch_bounds__(256) void ffn2_ln_kernel(
    const _Float16* __restrict__ ff1, const _Float16* __restrict__ W2T,
    const float* __restrict__ b2, const float* __restrict__ h,
    const float* __restrict__ g, const float* __restrict__ b, float* __restrict__ out) {
  __shared__ _Float16 As[16 * FF];   // 16KB
  __shared__ float hbuf[16 * HID];   // 8KB
  int tid = threadIdx.x, lane = tid & 31, wave = tid >> 5;
  int rowbase = blockIdx.x * 16;
  {
    const _Float16* src = ff1 + (size_t)rowbase * FF;
#pragma unroll
    for (int i = 0; i < 4; i++) {
      int c = (tid + 256 * i) * 8;  // 16B chunks
      async_lds_b128(&As[c], src + c);
    }
  }
  wait_async0();
  __syncthreads();
  int col = lane & 15, rb = (lane & 16) ? 8 : 0;
  int cg = wave * 16 + col;
  float bb = b2[cg];
  v8f acc;
#pragma unroll
  for (int r = 0; r < 8; r++) acc[r] = bb;
#pragma unroll
  for (int kc = 0; kc < 16; kc++)
    acc = WMMA_F16(ldsA_frag(As, lane, kc, FF), glbB_frag(W2T, cg, lane, kc, FF), acc);
#pragma unroll
  for (int r = 0; r < 8; r++) {
    int row = rb + r;
    hbuf[row * HID + cg] = acc[r] + h[(size_t)(rowbase + row) * HID + cg];
  }
  __syncthreads();
  if (tid < 16) {
    const float* hr = &hbuf[tid * HID];
    float mu = 0.0f;
    for (int i = 0; i < HID; i++) mu += hr[i];
    mu *= (1.0f / HID);
    float var = 0.0f;
    for (int i = 0; i < HID; i++) { float d = hr[i] - mu; var += d * d; }
    var *= (1.0f / HID);
    float rs = rsqrtf(var + 1e-5f);
    float* o = out + (size_t)(rowbase + tid) * HID;
    for (int i = 0; i < HID; i++) o[i] = (hr[i] - mu) * rs * g[i] + b[i];
  }
}

// ---------------------------------------------------------------------------
extern "C" void kernel_launch(void* const* d_in, const int* in_sizes, int n_in,
                              void* d_out, int out_size, void* d_ws, size_t ws_size,
                              hipStream_t stream) {
  const float* x    = (const float*)d_in[0];
  const float* Wq   = (const float*)d_in[1];
  const float* bq   = (const float*)d_in[2];
  const float* Wk   = (const float*)d_in[3];
  const float* bk   = (const float*)d_in[4];
  const float* Wv   = (const float*)d_in[5];
  const float* bv   = (const float*)d_in[6];
  const float* Wo   = (const float*)d_in[7];
  const float* bo   = (const float*)d_in[8];
  const float* ln1g = (const float*)d_in[9];
  const float* ln1b = (const float*)d_in[10];
  const float* W1   = (const float*)d_in[11];
  const float* b1   = (const float*)d_in[12];
  const float* W2   = (const float*)d_in[13];
  const float* b2   = (const float*)d_in[14];
  const float* ln2g = (const float*)d_in[15];
  const float* ln2b = (const float*)d_in[16];
  const int*   ei   = (const int*)d_in[17];
  float* out = (float*)d_out;

  // Workspace carve-out (256B aligned slices)
  char* p = (char*)d_ws;
  auto carve = [&](size_t bytes) -> void* {
    void* r = (void*)p;
    p += (bytes + 255) & ~(size_t)255;
    return r;
  };
  float*    Q     = (float*)carve((size_t)N_NODES * HID * 4);
  float*    Kb    = (float*)carve((size_t)N_NODES * HID * 4);
  float*    V     = (float*)carve((size_t)N_NODES * HID * 4);
  float*    agg   = (float*)carve((size_t)N_NODES * HID * 4);
  float*    hbuf  = (float*)carve((size_t)N_NODES * HID * 4);
  float*    nmax  = (float*)carve((size_t)N_NODES * HEADS * 4);
  float*    nsum  = (float*)carve((size_t)N_NODES * HEADS * 4);
  float*    esc   = (float*)carve((size_t)N_EDGES * HEADS * 4);
  _Float16* ff1   = (_Float16*)carve((size_t)N_NODES * FF * 2);
  _Float16* x16   = (_Float16*)carve((size_t)N_NODES * HID * 2);
  _Float16* h16   = (_Float16*)carve((size_t)N_NODES * HID * 2);
  _Float16* WqkvT = (_Float16*)carve((size_t)3 * HID * HID * 2);
  _Float16* WoT   = (_Float16*)carve((size_t)HID * HID * 2);
  _Float16* W1T   = (_Float16*)carve((size_t)HID * FF * 2);
  _Float16* W2T   = (_Float16*)carve((size_t)FF * HID * 2);

  // Weight prep (tiny) + f16 mirror of x
  transpose_f16_kernel<<<64, 256, 0, stream>>>(WqkvT,                 Wq, HID, HID);
  transpose_f16_kernel<<<64, 256, 0, stream>>>(WqkvT + HID * HID,     Wk, HID, HID);
  transpose_f16_kernel<<<64, 256, 0, stream>>>(WqkvT + 2 * HID * HID, Wv, HID, HID);
  transpose_f16_kernel<<<64, 256, 0, stream>>>(WoT, Wo, HID, HID);
  transpose_f16_kernel<<<128, 256, 0, stream>>>(W1T, W1, HID, FF);
  transpose_f16_kernel<<<128, 256, 0, stream>>>(W2T, W2, FF, HID);
  convert_f16_kernel<<<1024, 256, 0, stream>>>(x16, x, N_NODES * HID);
  init_ws_kernel<<<512, 256, 0, stream>>>(agg, nmax, nsum);

  const int rowTiles = N_NODES / 16;               // 50000 = 16 * 3125
  const int ehBlocks = (N_EDGES * HEADS + 255) / 256;

  qkv_gemm_kernel<<<rowTiles, 256, 0, stream>>>(x16, WqkvT, bq, bk, bv, Q, Kb, V);
  attn_score_kernel<<<ehBlocks, 256, 0, stream>>>(Q, Kb, ei, esc, nmax);
  attn_exp_kernel<<<ehBlocks, 256, 0, stream>>>(ei, esc, nmax, nsum);
  attn_agg_kernel<<<ehBlocks, 256, 0, stream>>>(ei, esc, nsum, V, agg);
  o_ln_kernel<<<rowTiles, 256, 0, stream>>>(agg, WoT, bo, x, ln1g, ln1b, hbuf, h16);
  ffn1_kernel<<<rowTiles, 256, 0, stream>>>(h16, W1T, b1, ff1);
  ffn2_ln_kernel<<<rowTiles, 256, 0, stream>>>(ff1, W2T, b2, hbuf, ln2g, ln2b, out);
}
